// NeuralMemoryODE_89687507075205
// MI455X (gfx1250) — matrix operand order repaired and checked by reading, not compiled, not run on hardware
//
#include <hip/hip_runtime.h>
#include <cstdint>
#include <cstddef>

// ---------------------------------------------------------------------------
// MI455X / gfx1250: bf16 WMMA GEMM (f32 accum) + fused RK4 ODE epilogue.
// Data staged to LDS with global_load_async_to_lds_b128 (ASYNCcnt path).
// ---------------------------------------------------------------------------

typedef __attribute__((ext_vector_type(16))) __bf16 v16bf;
typedef __attribute__((ext_vector_type(8)))  __bf16 v8bf;
typedef __attribute__((ext_vector_type(8)))  float  v8f;

#define ODE_DT (1.0f / 9.0f)

// --- CDNA5 async copy: global -> LDS, 16B per lane, tracked by ASYNCcnt -----
__device__ __forceinline__ void async_ld_b128(uint32_t lds_addr, uint64_t gbase,
                                              uint32_t goff) {
  // GVS mode: saddr(64b SGPR pair) + vaddr(32b byte offset); VDST = LDS addr.
  asm volatile("global_load_async_to_lds_b128 %0, %1, %2"
               :: "v"(lds_addr), "v"(goff), "s"(gbase)
               : "memory");
}
__device__ __forceinline__ void wait_async0() {
  asm volatile("s_wait_asynccnt 0" ::: "memory");
}

// Stage one 128(rows) x 32(bf16) tile = 128 x 64B = 512 x 16B chunks.
// 256 threads -> 2 async chunks each. LDS layout: row-major, 64B per row.
__device__ __forceinline__ void stage128x32(uint32_t ldsTile, const __bf16* g,
                                            uint32_t rowStrideB, uint32_t baseRow,
                                            uint32_t kkByte, uint32_t t) {
#pragma unroll
  for (int r = 0; r < 2; ++r) {
    uint32_t ch  = t + (uint32_t)r * 256u;
    uint32_t row = ch >> 2;
    uint32_t cb  = (ch & 3u) * 16u;
    uint32_t goff = (baseRow + row) * rowStrideB + kkByte + cb;
    async_ld_b128(ldsTile + ch * 16u, (uint64_t)(uintptr_t)g, goff);
  }
}

// dy/dt = -y + sin(y+g)^2 + exp(-y)*sin(y+g)^2
__device__ __forceinline__ float fode(float y, float g) {
  float s  = __sinf(y + g);
  float s2 = s * s;
  return __fmaf_rn(__expf(-y), s2, s2 - y);
}

// ---------------------------------------------------------------------------
// GEMM: C[128x128 per block] = A[M x K] * B^T[N x K]  (both bf16, K-major)
// Wave tile 32(M) x 64(N): 2x4 WMMA 16x16 f32 accumulators.
// FUSE_ODE: epilogue runs 9 RK4 steps elementwise, stores bf16 y_end.
// else:     epilogue adds bias, stores fp32 (n < Nout guard, ld = Nout).
// ---------------------------------------------------------------------------
template <bool FUSE_ODE>
__global__ __launch_bounds__(256, 1)
void wmma_gemm(const __bf16* __restrict__ A, const __bf16* __restrict__ B,
               const float* __restrict__ bias, float* __restrict__ outF,
               __bf16* __restrict__ outBF, int K, int Nout, int ldOut) {
  __shared__ __align__(64) __bf16 sA[2][4096];   // 2 x 8KB
  __shared__ __align__(64) __bf16 sB[2][4096];   // 2 x 8KB

  const uint32_t t  = threadIdx.x;
  const uint32_t w  = t >> 5;       // wave id 0..7 (wave32)
  const uint32_t l  = t & 31u;
  const uint32_t lr = l & 15u;      // lane row/col within 16
  const uint32_t kg = l >> 4;       // half-wave K group
  const uint32_t mBase = blockIdx.y * 128u;
  const uint32_t nBase = blockIdx.x * 128u;
  const uint32_t wm = (w >> 1) * 32u;   // wave M offset (4 waves down)
  const uint32_t wn = (w & 1u) * 64u;   // wave N offset (2 waves across)
  const uint32_t rowB = (uint32_t)K * 2u;  // K-major row stride in bytes

  uint32_t aL[2] = { (uint32_t)(uintptr_t)&sA[0][0], (uint32_t)(uintptr_t)&sA[1][0] };
  uint32_t bL[2] = { (uint32_t)(uintptr_t)&sB[0][0], (uint32_t)(uintptr_t)&sB[1][0] };

  v8f acc[2][4];
#pragma unroll
  for (int mi = 0; mi < 2; ++mi)
#pragma unroll
    for (int ni = 0; ni < 4; ++ni) acc[mi][ni] = v8f{};

  __builtin_prefetch(bias + nBase + wn + lr, 0, 0);   // global_prefetch_b8

  // Prologue: stage K-chunk 0 into buffer 0.
  stage128x32(aL[0], A, rowB, mBase, 0u, t);
  stage128x32(bL[0], B, rowB, nBase, 0u, t);
  wait_async0();
  __syncthreads();

  const int KT = K >> 5;   // chunks of 32 (bf16 WMMA K)
  for (int kt = 0; kt < KT; ++kt) {
    const int cur = kt & 1;
    if (kt + 1 < KT) {  // prefetch next chunk into the other buffer
      stage128x32(aL[cur ^ 1], A, rowB, mBase, (uint32_t)(kt + 1) * 64u, t);
      stage128x32(bL[cur ^ 1], B, rowB, nBase, (uint32_t)(kt + 1) * 64u, t);
    }
    const __bf16* As = &sA[cur][0];
    const __bf16* Bs = &sB[cur][0];

    // A frag 16x32 (ISA layout): VGPR0-3 = K[kg*8 .. kg*8+7], VGPR4-7 = K[16+kg*8 ..]
    v16bf af[2];
#pragma unroll
    for (int mi = 0; mi < 2; ++mi) {
      uint32_t row = wm + (uint32_t)mi * 16u + lr;
      union { v16bf v; v8bf h[2]; } u;
      u.h[0] = *(const v8bf*)(As + row * 32u + kg * 8u);
      u.h[1] = *(const v8bf*)(As + row * 32u + 16u + kg * 8u);
      af[mi] = u.v;
    }
    // B frag 32x16: same K order as A: K[kg*8 .. kg*8+7], then K[16+kg*8 ..] of column lr
    // (B is stored transposed / K-major).
    v16bf bfm[4];
#pragma unroll
    for (int ni = 0; ni < 4; ++ni) {
      uint32_t col = wn + (uint32_t)ni * 16u + lr;
      union { v16bf v; v8bf h[2]; } ub;
      ub.h[0] = *(const v8bf*)(Bs + col * 32u + kg * 8u);
      ub.h[1] = *(const v8bf*)(Bs + col * 32u + 16u + kg * 8u);
      bfm[ni] = ub.v;
    }

#pragma unroll
    for (int mi = 0; mi < 2; ++mi)
#pragma unroll
      for (int ni = 0; ni < 4; ++ni)
        acc[mi][ni] = __builtin_amdgcn_wmma_f32_16x16x32_bf16(
            false, af[mi], false, bfm[ni], (short)0, acc[mi][ni], false, false);

    wait_async0();     // our async stores into next buffer have landed in LDS
    __syncthreads();   // everyone done reading cur + done staging next
  }

  // -------------------------- epilogue ------------------------------------
#pragma unroll
  for (int mi = 0; mi < 2; ++mi) {
#pragma unroll
    for (int ni = 0; ni < 4; ++ni) {
      uint32_t n = nBase + wn + (uint32_t)ni * 16u + lr;
      float bn = 0.f;
      if (FUSE_ODE)            bn = bias[n];
      else if ((int)n < Nout)  bn = bias[n];
#pragma unroll
      for (int e = 0; e < 8; ++e) {
        // C/D layout: VGPR e -> M = e (lanes 0-15) / e+8 (lanes 16-31)
        uint32_t m = mBase + wm + (uint32_t)mi * 16u + (uint32_t)e + kg * 8u;
        float v = acc[mi][ni][e] + bn;
        if (FUSE_ODE) {
          // gamma stays in registers: run the whole 9-step RK4 here.
          float g = v, y = 0.f;
#pragma unroll 1
          for (int s = 0; s < 9; ++s) {
            float k1 = fode(y, g);
            float k2 = fode(__fmaf_rn(0.5f * ODE_DT, k1, y), g);
            float k3 = fode(__fmaf_rn(0.5f * ODE_DT, k2, y), g);
            float k4 = fode(__fmaf_rn(ODE_DT, k3, y), g);
            y = __fmaf_rn(ODE_DT * (1.0f / 6.0f),
                          k1 + 2.0f * k2 + 2.0f * k3 + k4, y);
          }
          outBF[(size_t)m * (size_t)ldOut + n] = (__bf16)y;
        } else {
          if ((int)n < Nout)
            outF[(size_t)m * (size_t)Nout + n] = v;
        }
      }
    }
  }
}

// --------------------- fp32 -> bf16 conversion kernels ----------------------
__global__ void cvt_bf16(const float* __restrict__ in, __bf16* __restrict__ out,
                         int n) {
  int i = blockIdx.x * blockDim.x + threadIdx.x;
  if (i < n) out[i] = (__bf16)in[i];
}

// WT[NP][K] = W[K][N]^T, zero-padded rows for n >= N (K-major output).
__global__ void cvtT_bf16(const float* __restrict__ W, __bf16* __restrict__ WT,
                          int K, int N, int NP) {
  int i = blockIdx.x * blockDim.x + threadIdx.x;
  if (i < NP * K) {
    int n = i / K, k = i - n * K;
    WT[i] = (n < N) ? (__bf16)W[(size_t)k * (size_t)N + (size_t)n] : (__bf16)0.f;
  }
}

// ---------------------------------------------------------------------------
extern "C" void kernel_launch(void* const* d_in, const int* in_sizes, int n_in,
                              void* d_out, int out_size, void* d_ws, size_t ws_size,
                              hipStream_t stream) {
  (void)in_sizes; (void)n_in; (void)out_size; (void)ws_size;
  const float* x     = (const float*)d_in[0];   // [8192,1024]
  const float* W_enc = (const float*)d_in[1];   // [1024,2048]
  const float* b_enc = (const float*)d_in[2];   // [2048]
  const float* W_cls = (const float*)d_in[3];   // [2048,1000]
  const float* b_cls = (const float*)d_in[4];   // [1000]
  float* out = (float*)d_out;                   // [8192,1000]

  constexpr int Bm = 8192, Din = 1024, H = 2048, Cn = 1000, Cp = 1024;

  char* ws = (char*)d_ws;
  __bf16* xb  = (__bf16*)(ws);                              // 16 MB  [8192][1024]
  __bf16* WeT = (__bf16*)(ws + (((size_t)16) << 20));       //  4 MB  [2048][1024]
  __bf16* WcT = (__bf16*)(ws + (((size_t)20) << 20));       //  4 MB  [1024][2048]
  __bf16* yb  = (__bf16*)(ws + (((size_t)24) << 20));       // 32 MB  [8192][2048]

  cvt_bf16<<<(Bm * Din) / 256, 256, 0, stream>>>(x, xb, Bm * Din);
  cvtT_bf16<<<(H * Din) / 256, 256, 0, stream>>>(W_enc, WeT, Din, H, H);
  cvtT_bf16<<<(Cp * H) / 256, 256, 0, stream>>>(W_cls, WcT, H, Cn, Cp);

  // GEMM1 + bias + RK4  -> y_end (bf16, ld=2048)
  wmma_gemm<true><<<dim3(H / 128, Bm / 128), 256, 0, stream>>>(
      xb, WeT, b_enc, nullptr, yb, Din, H, H);

  // GEMM2 + bias -> out (fp32, N padded to 1024, store guarded to 1000)
  wmma_gemm<false><<<dim3(Cp / 128, Bm / 128), 256, 0, stream>>>(
      yb, WcT, b_cls, out, nullptr, H, Cn, Cn);
}
